// feature_embedding_module_69157563400620
// MI455X (gfx1250) — matrix operand order repaired and checked by reading, not compiled, not run on hardware
//
#include <hip/hip_runtime.h>
#include <hip/hip_bf16.h>

// ---------------------------------------------------------------------------
// CDNA5 / gfx1250 implementation of Chai-1 feature embedding.
//  * one-hot groups  -> weight-column gather kernels (store-bandwidth bound)
//  * token (ESM)     -> v_wmma_f32_16x16x32_bf16 GEMM, 16x16 tile per wave32
//  * weights pre-transposed + bf16-rounded into d_ws by a prep kernel
// ---------------------------------------------------------------------------

typedef __attribute__((ext_vector_type(16))) __bf16 v16bf;
typedef __attribute__((ext_vector_type(8)))  float  v8f;

#define DEVINL static __device__ __forceinline__

DEVINL unsigned short bfbits(float f) {          // f32 -> bf16 bits, RNE
    union { float f; unsigned u; } v; v.f = f;
    unsigned u = v.u + 0x7FFFu + ((v.u >> 16) & 1u);
    return (unsigned short)(u >> 16);
}
DEVINL float bfr(float f) {                      // round to bf16 precision
    union { unsigned u; float f; } v; v.u = ((unsigned)bfbits(f)) << 16;
    return v.f;
}
DEVINL unsigned pk2(float a, float b) {
    return (unsigned)bfbits(a) | ((unsigned)bfbits(b) << 16);
}
DEVINL void add8(float* acc, const float* w, float s) {
    const float4* p = (const float4*)w;
    float4 w0 = p[0], w1 = p[1];
    acc[0] += s * w0.x; acc[1] += s * w0.y; acc[2] += s * w0.z; acc[3] += s * w0.w;
    acc[4] += s * w1.x; acc[5] += s * w1.y; acc[6] += s * w1.z; acc[7] += s * w1.w;
}
DEVINL void store8(unsigned short* op, const float* acc) {
    uint4 v;
    v.x = pk2(acc[0], acc[1]); v.y = pk2(acc[2], acc[3]);
    v.z = pk2(acc[4], acc[5]); v.w = pk2(acc[6], acc[7]);
    *(uint4*)op = v;
}

// ------------------------------- geometry ----------------------------------
constexpr int A_N = 11776;
constexpr int NT  = 512, M_MSA = 1024;
constexpr long long APAIR_ROWS = 368LL * 32 * 128;          // 1,507,328

constexpr long long OUT_ATOM = 0;
constexpr long long OUT_AP   = OUT_ATOM + (long long)A_N * 256;   //   3,014,656
constexpr long long OUT_TOK  = OUT_AP   + APAIR_ROWS * 32;        //  51,249,152
constexpr long long OUT_TP   = OUT_TOK  + (long long)NT * 384;    //  51,445,760
constexpr long long OUT_MSA  = OUT_TP   + (long long)NT * NT * 256;// 118,554,624
constexpr long long OUT_TMPL = OUT_MSA  + (long long)M_MSA * NT * 64; // 152,109,056

// workspace layout (float offsets); total 1,127,040 floats = 4.51 MB
constexpr int WS_ATOM = 0;                      // [395][256] W_atom^T
constexpr int WS_AP   = WS_ATOM + 395 * 256;    // [14][32]
constexpr int WS_TP   = WS_AP   + 14  * 32;     // [128][256]
constexpr int WS_MSA  = WS_TP   + 128 * 256;    // [43][64]
constexpr int WS_TMPL = WS_MSA  + 43  * 64;     // [75][64]
constexpr int WS_T2   = WS_TMPL + 75  * 64;     // [33][64]  emb @ W_tmpl[:,40:72]^T
constexpr int WS_ESM  = WS_T2   + 33  * 64;     // [384][2560] W_token ESM block
constexpr int WS_TOTAL = WS_ESM + 384 * 2560;

// ------------------------------ prep kernel --------------------------------
// Transpose every weight matrix (rounded to bf16 values), fold the template
// restype embedding into a 33x64 table, and extract the aligned ESM weight
// block for the WMMA B-operand.
__global__ __launch_bounds__(256)
void prep_kernel(const float* __restrict__ Wa,  const float* __restrict__ Wap,
                 const float* __restrict__ Wtp, const float* __restrict__ Wm,
                 const float* __restrict__ Wt,  const float* __restrict__ emb,
                 const float* __restrict__ Wtok, float* __restrict__ ws)
{
    int gid = blockIdx.x * 256 + threadIdx.x;
    if (gid < 395 * 256) { int c = gid >> 8, o = gid & 255; ws[WS_ATOM + gid] = bfr(Wa[o * 395 + c]); return; }
    gid -= 395 * 256;
    if (gid < 14 * 32)   { int c = gid >> 5, o = gid & 31;  ws[WS_AP   + gid] = bfr(Wap[o * 14 + c]); return; }
    gid -= 14 * 32;
    if (gid < 128 * 256) { int c = gid >> 8, o = gid & 255; ws[WS_TP   + gid] = bfr(Wtp[o * 128 + c]); return; }
    gid -= 128 * 256;
    if (gid < 43 * 64)   { int c = gid >> 6, o = gid & 63;  ws[WS_MSA  + gid] = bfr(Wm[o * 43 + c]); return; }
    gid -= 43 * 64;
    if (gid < 75 * 64)   { int c = gid >> 6, o = gid & 63;  ws[WS_TMPL + gid] = bfr(Wt[o * 75 + c]); return; }
    gid -= 75 * 64;
    if (gid < 33 * 64) {
        int r = gid >> 6, o = gid & 63;
        float acc = 0.f;
        for (int k = 0; k < 32; ++k) acc += bfr(emb[r * 32 + k]) * bfr(Wt[o * 75 + 40 + k]);
        ws[WS_T2 + gid] = acc; return;
    }
    gid -= 33 * 64;
    if (gid < 384 * 2560) {
        int n = gid / 2560, k = gid - n * 2560;
        ws[WS_ESM + gid] = bfr(Wtok[n * 2638 + 1 + k]);   // ESM cols are 1..2560
    }
}

// ------------------------- TOKEN: WMMA bf16 GEMM ---------------------------
// out[512,384] = ESM[512,2560] @ Wesm^T  (+ sparse one-hot epilogue + bias).
// One wave32 owns a 16x16 tile; K marches in steps of 32 through
// v_wmma_f32_16x16x32_bf16. A/B packing uses contiguous K per lane half:
//   lane<16 : row m=lane,   K = ka..ka+15   (ka = k0)
//   lane>=16: row m=lane-16,K shifted by 16 (ka = k0+16)
//   B lane n<16 : col n, K = k0..k0+15 ; lane>=16: K = k0+16..k0+31
__global__ __launch_bounds__(128)
void token_wmma_kernel(const float* __restrict__ esm_in,
                       const float* __restrict__ wesm,      // ws+WS_ESM [384][2560]
                       const float* __restrict__ W,         // W_token [384][2638]
                       const float* __restrict__ bias,
                       const float* __restrict__ chainCrop,
                       const int*   __restrict__ isDist,
                       const float* __restrict__ delMean,
                       const float* __restrict__ msaProf,
                       const float* __restrict__ missing,
                       const int*   __restrict__ resType,
                       const int*   __restrict__ bfac,
                       const int*   __restrict__ plddt,
                       unsigned short* __restrict__ out)    // d_out + OUT_TOK
{
    int wave = (blockIdx.x * blockDim.x + threadIdx.x) >> 5;
    if (wave >= 32 * 24) return;                 // 512/16 x 384/16 tiles
    int lane = threadIdx.x & 31;
    int half = lane >> 4, lr = lane & 15;
    int m0 = (wave / 24) * 16;
    int n0 = (wave - (wave / 24) * 24) * 16;

    const float* aRow = esm_in + (long long)(m0 + lr) * 2560;
    const float* bRow = wesm   + (long long)(n0 + lr) * 2560;

    v8f c = {};
    for (int k0 = 0; k0 < 2560; k0 += 32) {
        int ka = k0 + half * 16;
        int kb = k0 + half * 16;
        __builtin_prefetch(aRow + ka + 128, 0, 1);   // global_prefetch_b8
        float4 a0 = *(const float4*)(aRow + ka);
        float4 a1 = *(const float4*)(aRow + ka + 4);
        float4 a2 = *(const float4*)(aRow + ka + 8);
        float4 a3 = *(const float4*)(aRow + ka + 12);
        float4 b0 = *(const float4*)(bRow + kb);
        float4 b1 = *(const float4*)(bRow + kb + 4);
        float4 b2 = *(const float4*)(bRow + kb + 8);
        float4 b3 = *(const float4*)(bRow + kb + 12);
        v16bf a, b;
        a[0]=(__bf16)a0.x; a[1]=(__bf16)a0.y; a[2]=(__bf16)a0.z; a[3]=(__bf16)a0.w;
        a[4]=(__bf16)a1.x; a[5]=(__bf16)a1.y; a[6]=(__bf16)a1.z; a[7]=(__bf16)a1.w;
        a[8]=(__bf16)a2.x; a[9]=(__bf16)a2.y; a[10]=(__bf16)a2.z; a[11]=(__bf16)a2.w;
        a[12]=(__bf16)a3.x; a[13]=(__bf16)a3.y; a[14]=(__bf16)a3.z; a[15]=(__bf16)a3.w;
        b[0]=(__bf16)b0.x; b[1]=(__bf16)b0.y; b[2]=(__bf16)b0.z; b[3]=(__bf16)b0.w;
        b[4]=(__bf16)b1.x; b[5]=(__bf16)b1.y; b[6]=(__bf16)b1.z; b[7]=(__bf16)b1.w;
        b[8]=(__bf16)b2.x; b[9]=(__bf16)b2.y; b[10]=(__bf16)b2.z; b[11]=(__bf16)b2.w;
        b[12]=(__bf16)b3.x; b[13]=(__bf16)b3.y; b[14]=(__bf16)b3.z; b[15]=(__bf16)b3.w;
        c = __builtin_amdgcn_wmma_f32_16x16x32_bf16(false, a, false, b,
                                                    (short)0, c, false, false);
    }

    // Epilogue: bias + all non-ESM feature columns (C layout: VGPR i holds
    // M = half*8+i, N = n0+lr).
    int col = n0 + lr;
    const float* wc = W + (long long)col * 2638;
    float bcol = bfr(bias[col]);
    #pragma unroll
    for (int i = 0; i < 8; ++i) {
        int r = m0 + half * 8 + i;
        float acc = c[i] + bcol;
        acc += bfr(chainCrop[r]) * bfr(wc[0]);
        acc += bfr(wc[2561 + isDist[r]]);
        acc += bfr(delMean[r]) * bfr(wc[2563]);
        const float* mp = msaProf + r * 33;
        for (int j = 0; j < 33; ++j) acc += bfr(mp[j]) * bfr(wc[2564 + j]);
        acc += bfr(missing[r]) * bfr(wc[2597]);
        acc += bfr(wc[2598 + resType[r]]);
        acc += bfr(wc[2631 + bfac[r]]);
        acc += bfr(wc[2634 + plddt[r]]);
        out[(long long)r * 384 + col] = bfbits(acc);
    }
}

// ------------------------------ ATOM (gather) ------------------------------
// wave per atom row; lane owns 8 consecutive outputs (b128 load/store).
__global__ __launch_bounds__(256)
void atom_kernel(const int* __restrict__ nameOH, const float* __restrict__ charge,
                 const int* __restrict__ elem,   const float* __restrict__ mask,
                 const float* __restrict__ pos,  const float* __restrict__ WT,
                 const float* __restrict__ bias, unsigned short* __restrict__ out)
{
    int wave = (blockIdx.x * blockDim.x + threadIdx.x) >> 5;
    if (wave >= A_N) return;
    int lane = threadIdx.x & 31;
    int row  = wave, o0 = lane * 8;
    float acc[8];
    #pragma unroll
    for (int j = 0; j < 8; ++j) acc[j] = bfr(bias[o0 + j]);
    const int* nm = nameOH + row * 4;
    add8(acc, WT + (0   + nm[0]) * 256 + o0, 1.f);
    add8(acc, WT + (65  + nm[1]) * 256 + o0, 1.f);
    add8(acc, WT + (130 + nm[2]) * 256 + o0, 1.f);
    add8(acc, WT + (195 + nm[3]) * 256 + o0, 1.f);
    add8(acc, WT + 260 * 256 + o0, bfr(charge[row]));
    add8(acc, WT + (261 + elem[row]) * 256 + o0, 1.f);
    add8(acc, WT + 391 * 256 + o0, bfr(mask[row]));
    const float* pr = pos + row * 3;
    add8(acc, WT + 392 * 256 + o0, bfr(pr[0]));
    add8(acc, WT + 393 * 256 + o0, bfr(pr[1]));
    add8(acc, WT + 394 * 256 + o0, bfr(pr[2]));
    store8(out + (long long)row * 256 + o0, acc);
}

// ---------------------------- ATOM_PAIR (gather) ---------------------------
// wave covers 8 rows; 4 lanes per row, 8 outputs each -> contiguous b128s.
__global__ __launch_bounds__(256)
void atom_pair_kernel(const int* __restrict__ distog, const float* __restrict__ inv,
                      const float* __restrict__ WT,   const float* __restrict__ bias,
                      unsigned short* __restrict__ out)
{
    long long wave = ((long long)blockIdx.x * blockDim.x + threadIdx.x) >> 5;
    long long rowBase = wave * 8;
    if (rowBase >= APAIR_ROWS) return;
    int lane = threadIdx.x & 31;
    long long row = rowBase + (lane >> 2);
    int o0 = (lane & 3) * 8;
    float acc[8];
    #pragma unroll
    for (int j = 0; j < 8; ++j) acc[j] = bfr(bias[o0 + j]);
    add8(acc, WT + distog[row] * 32 + o0, 1.f);
    add8(acc, WT + 12 * 32 + o0, bfr(inv[row * 2 + 0]));
    add8(acc, WT + 13 * 32 + o0, bfr(inv[row * 2 + 1]));
    store8(out + row * 32 + o0, acc);
}

// --------------------------- TOKEN_PAIR (gather) ---------------------------
// W^T (128x256 f32 = 128 KB) staged in LDS (320 KB/WGP). Wave per (i,j) pair,
// lane owns 8 consecutive outputs; 5 one-hot columns + up to 12 restraint
// columns + bias, then a single b128 store per lane.
extern __shared__ float tp_lds[];
__global__ __launch_bounds__(256)
void token_pair_kernel(const int* __restrict__ dock,  const int* __restrict__ rchain,
                       const int* __restrict__ rent,  const int* __restrict__ rseq,
                       const int* __restrict__ rtok,  const float* __restrict__ dres,
                       const float* __restrict__ pres, const float* __restrict__ radD,
                       const float* __restrict__ radP, const float* __restrict__ WT,
                       const float* __restrict__ bias, unsigned short* __restrict__ out,
                       int nWaves)
{
    float* lw = tp_lds;                // [128][256]
    float* lb = tp_lds + 128 * 256;    // [256]
    for (int i = threadIdx.x; i < 128 * 256; i += blockDim.x) lw[i] = WT[i];
    for (int i = threadIdx.x; i < 256; i += blockDim.x) lb[i] = bfr(bias[i]);
    __syncthreads();
    float rd[6], rp[6];
    #pragma unroll
    for (int r = 0; r < 6; ++r) { rd[r] = radD[r]; rp[r] = radP[r]; }
    int lane = threadIdx.x & 31;
    int gw = blockIdx.x * (blockDim.x >> 5) + (threadIdx.x >> 5);
    int o0 = lane * 8;
    for (long long p = gw; p < (long long)NT * NT; p += nWaves) {
        float acc[8];
        const float4* bp = (const float4*)(lb + o0);
        float4 b0 = bp[0], b1 = bp[1];
        acc[0]=b0.x; acc[1]=b0.y; acc[2]=b0.z; acc[3]=b0.w;
        acc[4]=b1.x; acc[5]=b1.y; acc[6]=b1.z; acc[7]=b1.w;
        add8(acc, lw + dock[p]         * 256 + o0, 1.f);
        add8(acc, lw + (7  + rchain[p]) * 256 + o0, 1.f);
        add8(acc, lw + (13 + rent[p])  * 256 + o0, 1.f);
        add8(acc, lw + (16 + rseq[p])  * 256 + o0, 1.f);
        add8(acc, lw + (82 + rtok[p])  * 256 + o0, 1.f);
        float d = dres[p];
        if (d >= 0.f) {
            #pragma unroll
            for (int r = 0; r < 6; ++r)
                if (d < rd[r]) add8(acc, lw + (116 + r) * 256 + o0, 1.f);
        }
        float q = pres[p];
        if (q >= 0.f) {
            #pragma unroll
            for (int r = 0; r < 6; ++r)
                if (q < rp[r]) add8(acc, lw + (122 + r) * 256 + o0, 1.f);
        }
        store8(out + p * 256 + o0, acc);
    }
}

// ------------------------------- MSA (gather) ------------------------------
// wave covers 4 cells; 8 lanes per cell, 8 outputs each.
__global__ __launch_bounds__(256)
void msa_kernel(const int* __restrict__ paired, const int* __restrict__ src,
                const float* __restrict__ delv, const int* __restrict__ hasdel,
                const int* __restrict__ oh,     const float* __restrict__ WT,
                const float* __restrict__ bias, unsigned short* __restrict__ out)
{
    __shared__ float lw[43 * 64];
    __shared__ float lb[64];
    for (int i = threadIdx.x; i < 43 * 64; i += blockDim.x) lw[i] = WT[i];
    if (threadIdx.x < 64) lb[threadIdx.x] = bfr(bias[threadIdx.x]);
    __syncthreads();
    int lane = threadIdx.x & 31;
    long long wave = ((long long)blockIdx.x * (blockDim.x >> 5)) + (threadIdx.x >> 5);
    long long cell = wave * 4 + (lane >> 3);
    if (cell >= (long long)M_MSA * NT) return;
    int o0 = (lane & 7) * 8;
    float acc[8];
    const float4* bp = (const float4*)(lb + o0);
    float4 b0 = bp[0], b1 = bp[1];
    acc[0]=b0.x; acc[1]=b0.y; acc[2]=b0.z; acc[3]=b0.w;
    acc[4]=b1.x; acc[5]=b1.y; acc[6]=b1.z; acc[7]=b1.w;
    add8(acc, lw + paired[cell]        * 64 + o0, 1.f);
    add8(acc, lw + (2 + src[cell])     * 64 + o0, 1.f);
    add8(acc, lw + 7                   * 64 + o0, bfr(delv[cell]));
    add8(acc, lw + (8 + hasdel[cell])  * 64 + o0, 1.f);
    add8(acc, lw + (10 + oh[cell])     * 64 + o0, 1.f);
    store8(out + cell * 64 + o0, acc);
}

// ---------------------------- TEMPLATES (gather) ---------------------------
// restype embedding pre-folded into T2[33][64]; remaining features are
// one-hot / scalar-scaled columns.
__global__ __launch_bounds__(256)
void templates_kernel(const int* __restrict__ distog, const float* __restrict__ mask,
                      const int* __restrict__ restype, const float* __restrict__ uv,
                      const float* __restrict__ WT,    const float* __restrict__ T2,
                      const float* __restrict__ bias,  unsigned short* __restrict__ out)
{
    __shared__ float lw[75 * 64];
    __shared__ float lt[33 * 64];
    __shared__ float lb[64];
    for (int i = threadIdx.x; i < 75 * 64; i += blockDim.x) lw[i] = WT[i];
    for (int i = threadIdx.x; i < 33 * 64; i += blockDim.x) lt[i] = T2[i];
    if (threadIdx.x < 64) lb[threadIdx.x] = bfr(bias[threadIdx.x]);
    __syncthreads();
    int lane = threadIdx.x & 31;
    long long wave = ((long long)blockIdx.x * (blockDim.x >> 5)) + (threadIdx.x >> 5);
    long long cell = wave * 4 + (lane >> 3);
    if (cell >= 4LL * NT * NT) return;
    int o0 = (lane & 7) * 8;
    float acc[8];
    const float4* bp = (const float4*)(lb + o0);
    float4 b0 = bp[0], b1 = bp[1];
    acc[0]=b0.x; acc[1]=b0.y; acc[2]=b0.z; acc[3]=b0.w;
    acc[4]=b1.x; acc[5]=b1.y; acc[6]=b1.z; acc[7]=b1.w;
    add8(acc, lw + distog[cell] * 64 + o0, 1.f);
    add8(acc, lw + 39 * 64 + o0, bfr(mask[cell]));
    add8(acc, lt + restype[cell] * 64 + o0, 1.f);
    const float* u = uv + cell * 3;
    add8(acc, lw + 72 * 64 + o0, bfr(u[0]));
    add8(acc, lw + 73 * 64 + o0, bfr(u[1]));
    add8(acc, lw + 74 * 64 + o0, bfr(u[2]));
    store8(out + cell * 64 + o0, acc);
}

// -------------------------------- launcher ---------------------------------
extern "C" void kernel_launch(void* const* d_in, const int* in_sizes, int n_in,
                              void* d_out, int out_size, void* d_ws, size_t ws_size,
                              hipStream_t stream)
{
    (void)in_sizes; (void)n_in; (void)out_size; (void)ws_size;
    const int*   AtomNameOneHot  = (const int*)  d_in[0];
    const float* AtomRefCharge   = (const float*)d_in[1];
    const int*   AtomRefElement  = (const int*)  d_in[2];
    const float* AtomRefMask     = (const float*)d_in[3];
    const float* AtomRefPos      = (const float*)d_in[4];
    const int*   BAPDistogram    = (const int*)  d_in[5];
    const float* InvSqBAPD       = (const float*)d_in[6];
    const float* ChainIsCropped  = (const float*)d_in[7];
    const float* ESM             = (const float*)d_in[8];
    const int*   IsDistillation  = (const int*)  d_in[9];
    const float* MSADeletionMean = (const float*)d_in[10];
    const float* MSAProfile      = (const float*)d_in[11];
    const float* MissingChain    = (const float*)d_in[12];
    const int*   ResidueType     = (const int*)  d_in[13];
    const int*   TokenBFactor    = (const int*)  d_in[14];
    const int*   TokenPLDDT      = (const int*)  d_in[15];
    const int*   IsPairedMSA     = (const int*)  d_in[16];
    const int*   MSADataSource   = (const int*)  d_in[17];
    const float* MSADeletionVal  = (const float*)d_in[18];
    const int*   MSAHasDeletion  = (const int*)  d_in[19];
    const int*   MSAOneHot       = (const int*)  d_in[20];
    const int*   Dock            = (const int*)  d_in[21];
    const int*   RelChain        = (const int*)  d_in[22];
    const int*   RelEntity       = (const int*)  d_in[23];
    const int*   RelSeqSep       = (const int*)  d_in[24];
    const int*   RelTokSep       = (const int*)  d_in[25];
    const float* TokDistRes      = (const float*)d_in[26];
    const float* TokPocketRes    = (const float*)d_in[27];
    const int*   TmplDistogram   = (const int*)  d_in[28];
    const float* TmplMask        = (const float*)d_in[29];
    const int*   TmplResType     = (const int*)  d_in[30];
    const float* TmplUnitVec     = (const float*)d_in[31];
    const float* W_atom          = (const float*)d_in[32];
    const float* b_atom          = (const float*)d_in[33];
    const float* W_atom_pair     = (const float*)d_in[34];
    const float* b_atom_pair     = (const float*)d_in[35];
    const float* W_token         = (const float*)d_in[36];
    const float* b_token         = (const float*)d_in[37];
    const float* W_token_pair    = (const float*)d_in[38];
    const float* b_token_pair    = (const float*)d_in[39];
    const float* W_msa           = (const float*)d_in[40];
    const float* b_msa           = (const float*)d_in[41];
    const float* W_templates     = (const float*)d_in[42];
    const float* b_templates     = (const float*)d_in[43];
    const float* restype_emb     = (const float*)d_in[44];
    const float* radii_dist      = (const float*)d_in[45];
    const float* radii_pocket    = (const float*)d_in[46];

    float* ws = (float*)d_ws;
    unsigned short* out = (unsigned short*)d_out;

    // 1) weight transpose / folding (requires ~4.51 MB of d_ws)
    prep_kernel<<<(WS_TOTAL + 255) / 256, 256, 0, stream>>>(
        W_atom, W_atom_pair, W_token_pair, W_msa, W_templates,
        restype_emb, W_token, ws);

    // 2) token: WMMA GEMM over ESM + sparse epilogue (768 wave-tiles)
    token_wmma_kernel<<<192, 128, 0, stream>>>(
        ESM, ws + WS_ESM, W_token, b_token,
        ChainIsCropped, IsDistillation, MSADeletionMean, MSAProfile,
        MissingChain, ResidueType, TokenBFactor, TokenPLDDT,
        out + OUT_TOK);

    // 3) gather kernels (store-bandwidth bound)
    atom_kernel<<<(A_N * 32 + 255) / 256, 256, 0, stream>>>(
        AtomNameOneHot, AtomRefCharge, AtomRefElement, AtomRefMask, AtomRefPos,
        ws + WS_ATOM, b_atom, out + OUT_ATOM);

    atom_pair_kernel<<<(int)((APAIR_ROWS / 8) * 32 / 256), 256, 0, stream>>>(
        BAPDistogram, InvSqBAPD, ws + WS_AP, b_atom_pair, out + OUT_AP);

    size_t tpShm = (size_t)(128 * 256 + 256) * sizeof(float);  // 132 KB LDS
    token_pair_kernel<<<2048, 256, tpShm, stream>>>(
        Dock, RelChain, RelEntity, RelSeqSep, RelTokSep,
        TokDistRes, TokPocketRes, radii_dist, radii_pocket,
        ws + WS_TP, b_token_pair, out + OUT_TP, 2048 * 8);

    msa_kernel<<<16384, 256, 0, stream>>>(
        IsPairedMSA, MSADataSource, MSADeletionVal, MSAHasDeletion, MSAOneHot,
        ws + WS_MSA, b_msa, out + OUT_MSA);

    templates_kernel<<<32768, 256, 0, stream>>>(
        TmplDistogram, TmplMask, TmplResType, TmplUnitVec,
        ws + WS_TMPL, ws + WS_T2, b_templates, out + OUT_TMPL);
}
